// QRNN_35991825940562
// MI455X (gfx1250) — matrix operand order repaired and checked by reading, compiled, not run on hardware
//
#include <hip/hip_runtime.h>

// Problem dims (compile-time constants from the reference)
#define Tt   512
#define Bb   64
#define INd  512
#define Hd   1024
#define OUTd 512

typedef __attribute__((ext_vector_type(16))) __bf16 v16bf;
typedef __attribute__((ext_vector_type(8)))  float  v8f;

__device__ __forceinline__ unsigned short f2bf(float f) {
    unsigned int u = __builtin_bit_cast(unsigned int, f);
    u += 0x7FFFu + ((u >> 16) & 1u);            // round-to-nearest-even
    return (unsigned short)(u >> 16);
}
__device__ __forceinline__ float bf2f(unsigned short h) {
    unsigned int u = ((unsigned int)h) << 16;
    return __builtin_bit_cast(float, u);
}

// LDS byte offset of a generic pointer to __shared__ (aperture rule: addr[31:0])
__device__ __forceinline__ unsigned lds_off_of(const void* p) {
    return (unsigned)(unsigned long long)(uintptr_t)p;
}

// CDNA5 async memory->LDS copy, 16 bytes per lane, tracked by ASYNCcnt.
__device__ __forceinline__ void async_copy_b128(unsigned lds_byte_off,
                                                const void* gptr) {
    asm volatile("global_load_async_to_lds_b128 %0, %1, off"
                 :: "v"(lds_byte_off),
                    "v"((unsigned long long)(uintptr_t)gptr)
                 : "memory");
}
__device__ __forceinline__ void wait_async_all() {
    asm volatile("s_wait_asynccnt 0x0" ::: "memory");
}

// ---------------------------------------------------------------- converts
__global__ void qrnn_cvt_bf16(const float* __restrict__ in,
                              unsigned short* __restrict__ out, int n) {
    for (int i = blockIdx.x * blockDim.x + threadIdx.x; i < n;
         i += gridDim.x * blockDim.x)
        out[i] = f2bf(in[i]);
}

// Convert + transpose: in (K,N) f32 row-major -> out (N,K) bf16 row-major.
// One-time cost; column reads are L2-resident (weights << 192MB L2).
__global__ void qrnn_cvt_bf16_T(const float* __restrict__ in,
                                unsigned short* __restrict__ out,
                                int K, int N) {
    long total = (long)K * N;
    for (long i = blockIdx.x * blockDim.x + threadIdx.x; i < total;
         i += (long)gridDim.x * blockDim.x) {
        int n = (int)(i / K);
        int k = (int)(i % K);
        out[i] = f2bf(in[(long)k * N + n]);
    }
}

// ---------------------------------------------------------------- GEMM (WMMA bf16)
// C = act(A @ W + bias). A: (M,K) bf16 row-major. Bt: (N,K) bf16 row-major
// (pre-transposed weights). mode 0: gates epilogue (col<Hsplit -> tanh,
// else sigmoid), bf16 out. mode 1: plain bias add, f32 out.
#define BM 128
#define BN 128
#define BK 32
#define LDP 40   // padded LDS row stride (bf16 elems); 80B rows keep 16B align

__global__ __launch_bounds__(256)
void qrnn_gemm_wmma(const unsigned short* __restrict__ A,
                    const unsigned short* __restrict__ Bt,
                    const float* __restrict__ bias,
                    void* __restrict__ Cout,
                    int M, int N, int K, int mode, int Hsplit) {
    // double-buffered tiles: A 128x32, Bt 128x32 (both row-major over K)
    __shared__ unsigned short As[2][BM * LDP];
    __shared__ unsigned short Bs[2][BN * LDP];

    const int tid   = threadIdx.x;
    const int lane  = tid & 31;
    const int wave  = tid >> 5;       // 0..7
    const int waveM = wave >> 2;      // 0..1 -> 64-row strip
    const int waveN = wave & 3;       // 0..3 -> 32-col strip
    const int half  = lane >> 4;      // lane half (K split per ISA layout)
    const int l16   = lane & 15;

    const int nBase = blockIdx.x * BN;
    const int mBase = blockIdx.y * BM;

    // per-thread staging chunk: 16B (8 bf16); 512 chunks per tile, 2/thread
    const int crow = tid >> 2;        // 0..63  (+64 for second chunk)
    const int cseg = (tid & 3) * 8;   // bf16 element offset within row

    const size_t aOff0 = (size_t)(mBase + crow) * K + cseg;
    const size_t aOff1 = aOff0 + (size_t)64 * K;
    const size_t bOff0 = (size_t)(nBase + crow) * K + cseg;
    const size_t bOff1 = bOff0 + (size_t)64 * K;

    unsigned ldsA0[2], ldsB0[2];
    ldsA0[0] = lds_off_of(&As[0][crow * LDP + cseg]);
    ldsA0[1] = lds_off_of(&As[1][crow * LDP + cseg]);
    ldsB0[0] = lds_off_of(&Bs[0][crow * LDP + cseg]);
    ldsB0[1] = lds_off_of(&Bs[1][crow * LDP + cseg]);
    const unsigned ldsStep = 64u * LDP * 2u;   // 64 rows down, in bytes

    const v8f vzero = {0.f, 0.f, 0.f, 0.f, 0.f, 0.f, 0.f, 0.f};
    v8f acc[4][2];
#pragma unroll
    for (int i = 0; i < 4; ++i)
#pragma unroll
        for (int j = 0; j < 2; ++j) acc[i][j] = vzero;

    union Frag { v16bf v; unsigned int u[8]; };

    const int nk = K / BK;

    // prologue: stage tile 0 into buffer 0
    async_copy_b128(ldsA0[0],           A  + aOff0);
    async_copy_b128(ldsA0[0] + ldsStep, A  + aOff1);
    async_copy_b128(ldsB0[0],           Bt + bOff0);
    async_copy_b128(ldsB0[0] + ldsStep, Bt + bOff1);

    for (int kt = 0; kt < nk; ++kt) {
        const int cur = kt & 1;
        wait_async_all();      // this wave's copies for tile kt have landed
        __syncthreads();       // publish to all waves (and all reads of the
                               // other buffer are complete past this point)

        if (kt + 1 < nk) {     // fire-and-forget stage of tile kt+1
            const int k1  = (kt + 1) * BK;
            const int nb  = cur ^ 1;
            async_copy_b128(ldsA0[nb],           A  + aOff0 + k1);
            async_copy_b128(ldsA0[nb] + ldsStep, A  + aOff1 + k1);
            async_copy_b128(ldsB0[nb],           Bt + bOff0 + k1);
            async_copy_b128(ldsB0[nb] + ldsStep, Bt + bOff1 + k1);
        }

        // Build fragments per documented 16-bit VGPR layouts.
        Frag af[4], bfr[2];
#pragma unroll
        for (int mt = 0; mt < 4; ++mt) {
            const unsigned short* base =
                &As[cur][(waveM * 64 + mt * 16 + l16) * LDP];
#pragma unroll
            for (int p = 0; p < 8; ++p) {
                int kk = half * 8 + 2 * p + ((p >= 4) ? 8 : 0);
                af[mt].u[p] = *(const unsigned int*)(base + kk);
            }
        }
#pragma unroll
        for (int nt = 0; nt < 2; ++nt) {
            const unsigned short* base =
                &Bs[cur][(waveN * 32 + nt * 16 + l16) * LDP];
#pragma unroll
            for (int p = 0; p < 8; ++p) {
                int kk = half * 8 + 2 * p + ((p >= 4) ? 8 : 0);
                bfr[nt].u[p] = *(const unsigned int*)(base + kk);
            }
        }

#pragma unroll
        for (int mt = 0; mt < 4; ++mt)
#pragma unroll
            for (int nt = 0; nt < 2; ++nt)
                acc[mt][nt] = __builtin_amdgcn_wmma_f32_16x16x32_bf16(
                    false, af[mt].v, false, bfr[nt].v,
                    (short)0, acc[mt][nt], false, false);
    }

    // Epilogue: C/D layout -> lane col = l16 (+half picks rows 8..15)
#pragma unroll
    for (int mt = 0; mt < 4; ++mt) {
#pragma unroll
        for (int nt = 0; nt < 2; ++nt) {
            int col = nBase + waveN * 32 + nt * 16 + l16;
            float bv = bias[col];
#pragma unroll
            for (int r = 0; r < 8; ++r) {
                int row = mBase + waveM * 64 + mt * 16 + (half ? 8 + r : r);
                float v = acc[mt][nt][r] + bv;
                if (mode == 0) {
                    v = (col < Hsplit) ? tanhf(v)
                                       : (1.0f / (1.0f + __expf(-v)));
                    ((unsigned short*)Cout)[(size_t)row * N + col] = f2bf(v);
                } else {
                    ((float*)Cout)[(size_t)row * N + col] = v;
                }
            }
        }
    }
}

// ---------------------------------------------------------------- fo-pool scan
// gates: (T,B,3H) bf16 activated [z|f|o]; h: (T,B,H) bf16 = o * c
__global__ void qrnn_fo_pool(const unsigned short* __restrict__ g,
                             unsigned short* __restrict__ h) {
    int idx = blockIdx.x * blockDim.x + threadIdx.x;   // 0..B*H-1
    if (idx >= Bb * Hd) return;
    int b  = idx / Hd;
    int hh = idx % Hd;
    float c = 0.0f;
    for (int t = 0; t < Tt; ++t) {
        size_t base = ((size_t)t * Bb + b) * (3 * Hd) + hh;
        float z = bf2f(g[base]);
        float f = bf2f(g[base + Hd]);
        float o = bf2f(g[base + 2 * Hd]);
        c = f * c + (1.0f - f) * z;
        h[((size_t)t * Bb + b) * Hd + hh] = f2bf(o * c);
    }
}

// ---------------------------------------------------------------- launch
extern "C" void kernel_launch(void* const* d_in, const int* in_sizes, int n_in,
                              void* d_out, int out_size, void* d_ws, size_t ws_size,
                              hipStream_t stream) {
    const float* x   = (const float*)d_in[0];
    const float* W0  = (const float*)d_in[1];
    const float* b0  = (const float*)d_in[2];
    const float* W1  = (const float*)d_in[3];
    const float* b1  = (const float*)d_in[4];
    const float* Wfc = (const float*)d_in[5];
    const float* bfc = (const float*)d_in[6];

    char* ws = (char*)d_ws;
    size_t off = 0;
    auto carve = [&](size_t bytes) -> char* {
        char* p = ws + off;
        off += (bytes + 255) & ~(size_t)255;
        return p;
    };

    unsigned short* xb    = (unsigned short*)carve((size_t)Tt * Bb * INd * 2);
    unsigned short* w0t   = (unsigned short*)carve((size_t)INd * 3 * Hd * 2);
    unsigned short* w1t   = (unsigned short*)carve((size_t)Hd * 3 * Hd * 2);
    unsigned short* wfct  = (unsigned short*)carve((size_t)Hd * OUTd * 2);
    unsigned short* gbuf  = (unsigned short*)carve((size_t)Tt * Bb * 3 * Hd * 2);
    unsigned short* h0b   = (unsigned short*)carve((size_t)Tt * Bb * Hd * 2);
    unsigned short* h1b   = (unsigned short*)carve((size_t)Tt * Bb * Hd * 2);

    const int M = Tt * Bb;    // 32768

    qrnn_cvt_bf16<<<4096, 256, 0, stream>>>(x, xb, M * INd);
    // weights: convert + transpose to (N,K) so GEMM staging is row-contiguous
    qrnn_cvt_bf16_T<<<2048, 256, 0, stream>>>(W0,  w0t,  INd, 3 * Hd);
    qrnn_cvt_bf16_T<<<2048, 256, 0, stream>>>(W1,  w1t,  Hd,  3 * Hd);
    qrnn_cvt_bf16_T<<<1024, 256, 0, stream>>>(Wfc, wfct, Hd,  OUTd);

    dim3 blk(256);
    // Layer 0: gates = act(x @ W0 + b0)
    qrnn_gemm_wmma<<<dim3(3 * Hd / BN, M / BM), blk, 0, stream>>>(
        xb, w0t, b0, gbuf, M, 3 * Hd, INd, 0, Hd);
    qrnn_fo_pool<<<(Bb * Hd) / 256, 256, 0, stream>>>(gbuf, h0b);

    // Layer 1
    qrnn_gemm_wmma<<<dim3(3 * Hd / BN, M / BM), blk, 0, stream>>>(
        h0b, w1t, b1, gbuf, M, 3 * Hd, Hd, 0, Hd);
    qrnn_fo_pool<<<(Bb * Hd) / 256, 256, 0, stream>>>(gbuf, h1b);

    // FC: out = h1 @ Wfc + bfc   (f32 direct to d_out)
    qrnn_gemm_wmma<<<dim3(OUTd / BN, M / BM), blk, 0, stream>>>(
        h1b, wfct, bfc, d_out, M, OUTd, Hd, 1, 0);
}
